// CausalSelfAttention_81243601371723
// MI455X (gfx1250) — matrix operand order repaired and checked by reading, not compiled, not run on hardware
//
#include <hip/hip_runtime.h>
#include <hip/hip_bf16.h>
#include <cstdint>

// ---------------- problem constants (B=4, T=2048, C=2048, H=16, hd=128) ---
#define CDIM   2048
#define TLEN   2048
#define NBATCH 4
#define NHEADS 16
#define HD     128
#define NQKV   (3*CDIM)          // 6144
#define MTOT   (NBATCH*TLEN)     // 8192

typedef __attribute__((ext_vector_type(16))) __bf16 v16bf;
typedef __attribute__((ext_vector_type(8)))  float  v8f;
typedef __attribute__((ext_vector_type(4)))  unsigned int u32x4;
typedef __attribute__((ext_vector_type(8)))  int  i32x8;
typedef __attribute__((ext_vector_type(4)))  int  i32x4;

__device__ __forceinline__ unsigned short f2bf_u(float f) {
  union { float f; unsigned u; } v; v.f = f;
  unsigned r = v.u + 0x7FFFu + ((v.u >> 16) & 1u);   // round-to-nearest-even
  return (unsigned short)(r >> 16);
}

union Frag16 { v16bf v; uint4 u[2]; unsigned short s[16]; };

__device__ __forceinline__ v8f wmma_bf16(v16bf a, v16bf b, v8f c) {
  // D = A(16x32 bf16) * B(32x16 bf16) + C(16x16 f32)
  return __builtin_amdgcn_wmma_f32_16x16x32_bf16(false, a, false, b,
                                                 (short)0, c, false, false);
}

// ---------------------------------------------------------------------------
// Tensor Data Mover: 2D bf16 tile (tile_w elems x tile_h rows) global -> LDS.
// Optional LDS padding: pad_amount DWORDs inserted every 2^(pad_iv+1) DWORDs.
// Issued once (EXEC-independent DMA); caller waits TENSORcnt + barrier.
// ---------------------------------------------------------------------------
__device__ __forceinline__ void tdm_load_2d_bf16(
    unsigned lds_off, const unsigned short* gptr,
    unsigned tile_w, unsigned tile_h,
    unsigned tensor_w, unsigned tensor_h, unsigned stride_elems,
    unsigned pad_iv_code, unsigned pad_amt_code, bool pad_en)
{
#if defined(__AMDGCN__) && __has_builtin(__builtin_amdgcn_tensor_load_to_lds)
  unsigned long long ga = (unsigned long long)(uintptr_t)gptr;
  u32x4 g0;
  g0[0] = 1u;                                        // count=1, user mode
  g0[1] = lds_off;                                   // lds_addr
  g0[2] = (unsigned)(ga & 0xFFFFFFFFu);              // global_addr[31:0]
  g0[3] = (unsigned)((ga >> 32) & 0x1FFFFFFu) | (2u << 30); // [56:32] | type=2
  i32x8 g1;
  g1[0] = (int)((1u << 16)                           // data_size = 2 bytes
              | (pad_en ? (1u << 20) : 0u)
              | (pad_iv_code << 22) | (pad_amt_code << 25));
  g1[1] = (int)(tensor_w << 16);                     // tensor_dim0[15:0]
  g1[2] = (int)((tensor_w >> 16) | (tensor_h << 16));// dim0 hi | dim1 lo
  g1[3] = (int)((tensor_h >> 16) | (tile_w << 16));  // dim1 hi | tile_dim0
  g1[4] = (int)(tile_h & 0xFFFF);                    // tile_dim1 (tile_dim2=0)
  g1[5] = (int)stride_elems;                         // tensor_dim0_stride lo
  g1[6] = 0;                                         // stride hi | dim1_stride
  g1[7] = 0;
  i32x4 z4; z4[0] = z4[1] = z4[2] = z4[3] = 0;
#if __clang_major__ >= 23
  i32x8 z8;
  #pragma unroll
  for (int i = 0; i < 8; ++i) z8[i] = 0;
  __builtin_amdgcn_tensor_load_to_lds(g0, g1, z4, z4, z8, 0);
#else
  __builtin_amdgcn_tensor_load_to_lds(g0, g1, z4, z4, 0);
#endif
#endif
}

// ===========================================================================
// Kernel 1: qkv = x @ W_qkv (bf16 WMMA, f32 accum), fused RoPE epilogue.
// Tile BM=128,BN=64,BK=32; 256 thr = 8 waves; wave w owns rows w*16..+15.
// B tile stored TRANSPOSED in LDS so B-fragments load as 2x b128.
// ===========================================================================
__global__ __launch_bounds__(256)
void qkv_rope_kernel(const float* __restrict__ x,
                     const float* __restrict__ fcos,
                     const float* __restrict__ fsin,
                     const float* __restrict__ Wqkv,
                     float* __restrict__ k_out,
                     float* __restrict__ v_out,
                     unsigned short* __restrict__ Qws,
                     unsigned short* __restrict__ Kws,
                     unsigned short* __restrict__ Vtws)
{
  __shared__ __attribute__((aligned(16))) unsigned short As[128][40];  // [M][K]
  __shared__ __attribute__((aligned(16))) unsigned short BsT[64][40];  // [N][K]
  __shared__ float Cs[128*64];

  const int tid  = threadIdx.x;
  const int wv   = tid >> 5;
  const int lane = tid & 31;
  const int hf   = lane >> 4;
  const int l16  = lane & 15;

  const int m_base = blockIdx.y * 128;
  const int n_base = blockIdx.x * 64;

  v8f acc[4];
  #pragma unroll
  for (int j = 0; j < 4; ++j)
    #pragma unroll
    for (int r = 0; r < 8; ++r) acc[j][r] = 0.f;

  for (int kk = 0; kk < CDIM; kk += 32) {
    // --- stage A tile (128x32 f32 -> bf16) ---
    #pragma unroll
    for (int l = 0; l < 4; ++l) {
      int idx = tid + l * 256;                  // 0..1023
      int row = idx >> 3;
      int c4  = (idx & 7) << 2;
      float4 f4 = *reinterpret_cast<const float4*>(
          &x[(size_t)(m_base + row) * CDIM + kk + c4]);
      As[row][c4+0] = f2bf_u(f4.x);  As[row][c4+1] = f2bf_u(f4.y);
      As[row][c4+2] = f2bf_u(f4.z);  As[row][c4+3] = f2bf_u(f4.w);
    }
    // --- stage B tile (32x64 f32 -> bf16), transposed into [N][K] ---
    #pragma unroll
    for (int l = 0; l < 2; ++l) {
      int idx = tid + l * 256;                  // 0..511
      int row = idx >> 4;                       // k 0..31
      int c4  = (idx & 15) << 2;                // n 0,4..60
      float4 f4 = *reinterpret_cast<const float4*>(
          &Wqkv[(size_t)(kk + row) * NQKV + n_base + c4]);
      BsT[c4+0][row] = f2bf_u(f4.x);  BsT[c4+1][row] = f2bf_u(f4.y);
      BsT[c4+2][row] = f2bf_u(f4.z);  BsT[c4+3][row] = f2bf_u(f4.w);
    }
    __syncthreads();

    // A fragment: row m=l16, K = hf*8+{0..7} and 16+hf*8+{0..7} (2x b128)
    Frag16 fa;
    const int arow = wv * 16 + l16;
    fa.u[0] = *reinterpret_cast<const uint4*>(&As[arow][hf*8]);
    fa.u[1] = *reinterpret_cast<const uint4*>(&As[arow][16 + hf*8]);

    #pragma unroll
    for (int j = 0; j < 4; ++j) {
      // B fragment: col n=l16, K = hf*8+{0..7} and 16+hf*8+{0..7} (2x b128 from transposed tile)
      Frag16 fb;
      fb.u[0] = *reinterpret_cast<const uint4*>(&BsT[j*16 + l16][hf*8]);
      fb.u[1] = *reinterpret_cast<const uint4*>(&BsT[j*16 + l16][16 + hf*8]);
      acc[j] = wmma_bf16(fa.v, fb.v, acc[j]);
    }
    __syncthreads();
  }

  // --- stage C to LDS so RoPE pairs (adjacent columns) are addressable ---
  #pragma unroll
  for (int j = 0; j < 4; ++j)
    #pragma unroll
    for (int r = 0; r < 8; ++r)
      Cs[(wv*16 + hf*8 + r) * 64 + j*16 + l16] = acc[j][r];
  __syncthreads();

  // --- RoPE + scatter epilogue: 16 column-pairs per thread ---
  for (int l = 0; l < 16; ++l) {
    int e  = tid + l * 256;          // 0..4095
    int mr = e >> 5;                 // row in tile 0..127
    int pc = (e & 31) << 1;          // pair column 0,2,..,62
    int m  = m_base + mr;
    int n  = n_base + pc;
    int b  = m >> 11, t = m & (TLEN - 1);
    int sect = n / CDIM;             // 0=q 1=k 2=v
    int rem  = n % CDIM;
    int h = rem >> 7, d = rem & (HD - 1);

    float x0 = Cs[mr*64 + pc], x1 = Cs[mr*64 + pc + 1];
    float o0 = x0, o1 = x1;
    if (sect < 2) {
      float c = fcos[t*(HD/2) + (d >> 1)];
      float s = fsin[t*(HD/2) + (d >> 1)];
      o0 = x0 * c - x1 * s;
      o1 = x0 * s + x1 * c;
    }
    size_t hm = (((size_t)(b*NHEADS + h) * TLEN) + t) * HD + d;  // (B,H,T,hd)
    if (sect == 0) {
      Qws[hm] = f2bf_u(o0);  Qws[hm+1] = f2bf_u(o1);
    } else if (sect == 1) {
      k_out[hm] = o0;  k_out[hm+1] = o1;
      Kws[hm] = f2bf_u(o0);  Kws[hm+1] = f2bf_u(o1);
    } else {
      v_out[hm] = o0;  v_out[hm+1] = o1;
      size_t vt = (((size_t)(b*NHEADS + h) * HD) + d) * TLEN + t; // (B,H,hd,T)
      Vtws[vt]        = f2bf_u(o0);
      Vtws[vt + TLEN] = f2bf_u(o1);
    }
  }
}

// ===========================================================================
// Kernel 2: causal flash attention, bf16 WMMA. One wave per 16-query tile.
// 32-key blocks: QK^T = 8 wmma, PV = 8 wmma. P staged bf16 in LDS so the
// P A-fragment reloads as 2x b128 (no scalar repack).
// ===========================================================================
__global__ __launch_bounds__(256)
void flash_attn_kernel(const unsigned short* __restrict__ Qws,
                       const unsigned short* __restrict__ Kws,
                       const unsigned short* __restrict__ Vtws,
                       unsigned short* __restrict__ Yws)
{
  __shared__ __attribute__((aligned(16))) unsigned short Psb[8][16][40];

  const int tid  = threadIdx.x;
  const int wv   = tid >> 5;
  const int lane = tid & 31;
  const int hf   = lane >> 4;
  const int l16  = lane & 15;

  const int bh = blockIdx.y;               // 0..63
  const int b  = bh >> 4, h = bh & 15;
  const int qt = blockIdx.x * 8 + wv;      // query tile 0..127
  const int m0 = qt * 16;

  const unsigned short* Qp = Qws  + (size_t)bh * TLEN * HD;
  const unsigned short* Kp = Kws  + (size_t)bh * TLEN * HD;
  const unsigned short* Vp = Vtws + (size_t)bh * HD * TLEN;

  // Q fragments for all 4 hd-chunks of 32 (resident whole loop)
  Frag16 aq[4];
  {
    const int qrow = m0 + l16;
    #pragma unroll
    for (int c = 0; c < 4; ++c) {
      const uint4* p = reinterpret_cast<const uint4*>(
          Qp + (size_t)qrow * HD + c*32 + hf*8);
      aq[c].u[0] = p[0];          // K = hf*8 + 0..7
      aq[c].u[1] = p[2];          // K = 16 + hf*8 + 0..7
    }
  }

  float m_i[8], l_i[8];
  v8f acc[8];
  #pragma unroll
  for (int r = 0; r < 8; ++r) { m_i[r] = -1e30f; l_i[r] = 0.f; }
  #pragma unroll
  for (int j = 0; j < 8; ++j)
    #pragma unroll
    for (int r = 0; r < 8; ++r) acc[j][r] = 0.f;

  const int   kb_count = ((qt + 1) * 16 + 31) >> 5;
  const float sm_scale = 0.08838834764831845f;   // 1/sqrt(128)

  for (int kb = 0; kb < kb_count; ++kb) {
    const int key_base = kb * 32;

    // ---- scores: two 16-key C tiles ----
    v8f sc0, sc1;
    #pragma unroll
    for (int r = 0; r < 8; ++r) { sc0[r] = 0.f; sc1[r] = 0.f; }
    #pragma unroll
    for (int c = 0; c < 4; ++c) {
      Frag16 bk0, bk1;   // B frag: lane holds key n, K-dim d = c*32 + hf*8 + {0..7}, c*32 + 16 + hf*8 + {0..7}
      const uint4* p0 = reinterpret_cast<const uint4*>(
          Kp + (size_t)(key_base      + l16) * HD + c*32 + hf*8);
      const uint4* p1 = reinterpret_cast<const uint4*>(
          Kp + (size_t)(key_base + 16 + l16) * HD + c*32 + hf*8);
      bk0.u[0] = p0[0]; bk0.u[1] = p0[2];
      bk1.u[0] = p1[0]; bk1.u[1] = p1[2];
      sc0 = wmma_bf16(aq[c].v, bk0.v, sc0);
      sc1 = wmma_bf16(aq[c].v, bk1.v, sc1);
    }

    // ---- mask + online softmax (lanes = key columns) ----
    const int qrow = m0 + hf * 8;            // row for slot r is qrow + r
    float lm[8];
    #pragma unroll
    for (int r = 0; r < 8; ++r) {
      int row = qrow + r;
      float s0 = sc0[r] * sm_scale;
      float s1 = sc1[r] * sm_scale;
      if (key_base + l16      > row) s0 = -1e30f;
      if (key_base + 16 + l16 > row) s1 = -1e30f;
      sc0[r] = s0; sc1[r] = s1;
      lm[r] = fmaxf(s0, s1);
    }
    #pragma unroll
    for (int off = 1; off < 16; off <<= 1)
      #pragma unroll
      for (int r = 0; r < 8; ++r)
        lm[r] = fmaxf(lm[r], __shfl_xor(lm[r], off, 32));

    float scl[8], ps[8];
    #pragma unroll
    for (int r = 0; r < 8; ++r) {
      float mn = fmaxf(m_i[r], lm[r]);
      scl[r]  = __expf(m_i[r] - mn);
      sc0[r]  = __expf(sc0[r] - mn);         // reuse as p0
      sc1[r]  = __expf(sc1[r] - mn);         // reuse as p1
      m_i[r]  = mn;
      ps[r]   = sc0[r] + sc1[r];
    }
    #pragma unroll
    for (int off = 1; off < 16; off <<= 1)
      #pragma unroll
      for (int r = 0; r < 8; ++r)
        ps[r] += __shfl_xor(ps[r], off, 32);
    #pragma unroll
    for (int r = 0; r < 8; ++r) l_i[r] = l_i[r] * scl[r] + ps[r];

    // ---- transpose P (C-layout) -> A-layout via per-wave LDS, bf16 ----
    #pragma unroll
    for (int r = 0; r < 8; ++r) {
      Psb[wv][hf*8 + r][l16]      = f2bf_u(sc0[r]);
      Psb[wv][hf*8 + r][16 + l16] = f2bf_u(sc1[r]);
    }
    asm volatile("s_wait_dscnt 0" ::: "memory");  // same-wave DS is in-order
    Frag16 ap;
    ap.u[0] = *reinterpret_cast<const uint4*>(&Psb[wv][l16][hf*8]);
    ap.u[1] = *reinterpret_cast<const uint4*>(&Psb[wv][l16][16 + hf*8]);

    // ---- rescale accumulators, then PV ----
    #pragma unroll
    for (int j = 0; j < 8; ++j)
      #pragma unroll
      for (int r = 0; r < 8; ++r) acc[j][r] *= scl[r];

    #pragma unroll
    for (int j = 0; j < 8; ++j) {
      Frag16 bv;  // lane holds d = j*16+l16, K-dim key = key_base + hf*8 + {0..7}, key_base + 16 + hf*8 + {0..7}
      const uint4* pv = reinterpret_cast<const uint4*>(
          Vp + (size_t)(j*16 + l16) * TLEN + key_base + hf*8);
      bv.u[0] = pv[0]; bv.u[1] = pv[2];
      acc[j] = wmma_bf16(ap.v, bv.v, acc[j]);
    }
  }

  // ---- normalize and write bf16 Y in (B,T,C) layout for out-proj ----
  #pragma unroll
  for (int j = 0; j < 8; ++j)
    #pragma unroll
    for (int r = 0; r < 8; ++r) {
      int t = m0 + hf*8 + r;
      int d = j*16 + l16;
      float o = acc[j][r] / l_i[r];
      Yws[((size_t)(b*TLEN + t)) * CDIM + h*HD + d] = f2bf_u(o);
    }
}

// ===========================================================================
// Kernel 3: out = Y @ W_out (bf16 WMMA, f32 accum). The bf16 A tile is moved
// by the Tensor Data Mover (global->LDS DMA, TDM pad recreates [128][40]),
// with a manual-staging fallback if the builtin is unavailable.
// ===========================================================================
__global__ __launch_bounds__(256)
void out_proj_kernel(const unsigned short* __restrict__ Yws,
                     const float* __restrict__ Wout,
                     float* __restrict__ out)
{
  __shared__ __attribute__((aligned(16))) unsigned short As[128][40];  // [M][K]
  __shared__ __attribute__((aligned(16))) unsigned short BsT[64][40];  // [N][K]

  const int tid  = threadIdx.x;
  const int wv   = tid >> 5;
  const int lane = tid & 31;
  const int hf   = lane >> 4;
  const int l16  = lane & 15;

  const int m_base = blockIdx.y * 128;
  const int n_base = blockIdx.x * 64;

#if defined(__AMDGCN__) && __has_builtin(__builtin_amdgcn_tensor_load_to_lds)
  const unsigned as_lds = (unsigned)(uintptr_t)&As[0][0];  // LDS aperture: low
#endif                                                     // 32 bits = offset

  v8f acc[4];
  #pragma unroll
  for (int j = 0; j < 4; ++j)
    #pragma unroll
    for (int r = 0; r < 8; ++r) acc[j][r] = 0.f;

  for (int kk = 0; kk < CDIM; kk += 32) {
#if defined(__AMDGCN__) && __has_builtin(__builtin_amdgcn_tensor_load_to_lds)
    // --- TDM: 32x128-elem bf16 tile, row stride 2048 elems, LDS pad 4 DWORDs
    //     every 16 DWORDs -> exactly As[128][40]. One issue per workgroup. ---
    if (wv == 0) {
      tdm_load_2d_bf16(as_lds, Yws + (size_t)m_base * CDIM + kk,
                       /*tile_w=*/32, /*tile_h=*/128,
                       /*tensor_w=*/CDIM, /*tensor_h=*/MTOT,
                       /*stride=*/CDIM,
                       /*pad_iv: 1<<(3+1)=16 DW*/3, /*pad_amt: 4 DW*/3, true);
      __builtin_amdgcn_s_wait_tensorcnt(0);
    }
#else
    #pragma unroll
    for (int l = 0; l < 2; ++l) {
      int idx = tid + l * 256;               // 0..511
      int row = idx >> 2;
      int c8  = (idx & 3) << 3;
      union { uint4 q; unsigned short s[8]; } v;
      v.q = *reinterpret_cast<const uint4*>(
          &Yws[(size_t)(m_base + row) * CDIM + kk + c8]);
      #pragma unroll
      for (int i = 0; i < 8; ++i) As[row][c8 + i] = v.s[i];
    }
#endif
    // --- stage B tile (f32 -> bf16), transposed into [N][K] ---
    #pragma unroll
    for (int l = 0; l < 2; ++l) {
      int idx = tid + l * 256;
      int row = idx >> 4;                    // k 0..31
      int c4  = (idx & 15) << 2;             // n 0,4..60
      float4 f4 = *reinterpret_cast<const float4*>(
          &Wout[(size_t)(kk + row) * CDIM + n_base + c4]);
      BsT[c4+0][row] = f2bf_u(f4.x);  BsT[c4+1][row] = f2bf_u(f4.y);
      BsT[c4+2][row] = f2bf_u(f4.z);  BsT[c4+3][row] = f2bf_u(f4.w);
    }
    __syncthreads();

    Frag16 fa;
    const int arow = wv * 16 + l16;
    fa.u[0] = *reinterpret_cast<const uint4*>(&As[arow][hf*8]);
    fa.u[1] = *reinterpret_cast<const uint4*>(&As[arow][16 + hf*8]);

    #pragma unroll
    for (int j = 0; j < 4; ++j) {
      Frag16 fb;
      fb.u[0] = *reinterpret_cast<const uint4*>(&BsT[j*16 + l16][hf*8]);
      fb.u[1] = *reinterpret_cast<const uint4*>(&BsT[j*16 + l16][16 + hf*8]);
      acc[j] = wmma_bf16(fa.v, fb.v, acc[j]);
    }
    __syncthreads();
  }

  #pragma unroll
  for (int j = 0; j < 4; ++j)
    #pragma unroll
    for (int r = 0; r < 8; ++r)
      out[(size_t)(m_base + wv*16 + hf*8 + r) * CDIM + n_base + j*16 + l16] =
          acc[j][r];
}

// ===========================================================================
extern "C" void kernel_launch(void* const* d_in, const int* in_sizes, int n_in,
                              void* d_out, int out_size, void* d_ws, size_t ws_size,
                              hipStream_t stream) {
  const float* x    = (const float*)d_in[0];
  const float* fcos = (const float*)d_in[1];
  const float* fsin = (const float*)d_in[2];
  const float* Wqkv = (const float*)d_in[3];
  const float* Wout = (const float*)d_in[4];

  float* out = (float*)d_out;
  const size_t BTC = (size_t)NBATCH * TLEN * CDIM;   // 16,777,216
  float* k_out = out + BTC;        // (B,H,T,hd) f32
  float* v_out = out + 2 * BTC;    // (B,H,T,hd) f32

  unsigned short* Qws = (unsigned short*)d_ws;   // bf16 (B,H,T,hd)
  unsigned short* Kws = Qws + BTC;               // bf16 (B,H,T,hd)
  unsigned short* Vt  = Kws + BTC;               // bf16 (B,H,hd,T)
  unsigned short* Yws = Vt  + BTC;               // bf16 (B,T,C)

  dim3 g1(NQKV / 64, MTOT / 128);                // 96 x 64
  qkv_rope_kernel<<<g1, 256, 0, stream>>>(x, fcos, fsin, Wqkv,
                                          k_out, v_out, Qws, Kws, Vt);

  dim3 g2(TLEN / 128, NBATCH * NHEADS);          // 16 x 64
  flash_attn_kernel<<<g2, 256, 0, stream>>>(Qws, Kws, Vt, Yws);

  dim3 g3(CDIM / 64, MTOT / 128);                // 32 x 64
  out_proj_kernel<<<g3, 256, 0, stream>>>(Yws, Wout, out);
}
